// MultiHeadAttentionLayer_3547642987279
// MI455X (gfx1250) — matrix operand order repaired and checked by reading, not compiled, not run on hardware
//
#include <hip/hip_runtime.h>
#include <hip/hip_bf16.h>

// MHA: B=4, S=2048, D=1024, H=16, HD=64.  All matrix math via
// v_wmma_f32_16x16x32_bf16 (bf16 A/B, f32 accumulate).
#define BB  4
#define SS  2048
#define DD  1024
#define HH  16
#define HDD 64

__constant__ const float kScale = 0.125f;   // 1/sqrt(64)
__constant__ const float kNeg   = -1e9f;

typedef __attribute__((ext_vector_type(16))) __bf16 v16bf;
typedef __attribute__((ext_vector_type(8)))  float  v8f;

union Frag {
  v16bf bf;
  unsigned short u[16];
  unsigned int   w[8];
  uint4          q[2];
};

__device__ inline unsigned short f2bf(float f) {
  unsigned int x = __float_as_uint(f);
  x += 0x7FFFu + ((x >> 16) & 1u);           // round to nearest even
  return (unsigned short)(x >> 16);
}

__device__ inline unsigned int pkbf(float lo, float hi) {
  unsigned int a = __float_as_uint(lo);
  unsigned int b = __float_as_uint(hi);
  a += 0x7FFFu + ((a >> 16) & 1u);
  b += 0x7FFFu + ((b >> 16) & 1u);
  return (a >> 16) | (b & 0xFFFF0000u);
}

__device__ inline v8f zero8() {
  v8f z;
#pragma unroll
  for (int i = 0; i < 8; ++i) z[i] = 0.0f;
  return z;
}

__device__ inline v8f wmma_bf16(const Frag& a, const Frag& b, v8f c) {
  // 8 args: (neg_a, A, neg_b, B, c_mod, C, reuse_a, reuse_b)
  return __builtin_amdgcn_wmma_f32_16x16x32_bf16(
      false, a.bf, false, b.bf, (short)0, c, false, false);
}

// ---------------------------------------------------------------------------
// GEMM: C[M=8192, N=1024] = A[M,1024] @ W^T   (W row-major [N,K], nn.Linear)
// One wave (32 threads) computes a 32x64 tile: 2 M-tiles x 4 N-tiles.
//   AF32  : A source is f32 (convert to bf16 in-register) else bf16 (u16).
//   QKVOUT: store bf16 into [B,H,S,HD] layout; else store f32 into [M,N].
// Fragment layouts per CDNA5 ISA 7.12.2:
//   A (16x32 bf16): lanes 0-15 row M=lane, halves K={kh..kh+7, 16+kh..16+kh+7},
//                   kh = 8*(lane>=16).
//   B (32x16 bf16): lane holds column N=lane%16, K={kh..kh+7, 16+kh..16+kh+7}, same as A.
// ---------------------------------------------------------------------------
template <bool AF32, bool QKVOUT>
__global__ __launch_bounds__(32) void gemm_xwt(const void* __restrict__ Ain,
                                               const float* __restrict__ W,
                                               void* __restrict__ Out) {
  const int lane  = threadIdx.x & 31;
  const int lhalf = lane >> 4;
  const int l16   = lane & 15;
  const int n0    = blockIdx.x * 64;
  const int m0    = blockIdx.y * 32;

  v8f acc[2][4];
#pragma unroll
  for (int i = 0; i < 2; ++i)
#pragma unroll
    for (int j = 0; j < 4; ++j) acc[i][j] = zero8();

  const float*          Af = (const float*)Ain;
  const unsigned short* Ab = (const unsigned short*)Ain;

  for (int k0 = 0; k0 < DD; k0 += 32) {
    if (k0 + 32 < DD)  // global_prefetch_b8 of next W K-block
      __builtin_prefetch(W + (size_t)(n0 + l16) * DD + k0 + 32, 0, 1);

    Frag a[2];
#pragma unroll
    for (int mt = 0; mt < 2; ++mt) {
      const size_t row = (size_t)(m0 + mt * 16 + l16);
      const int    kh  = lhalf * 8;
      if constexpr (AF32) {
        const float* p = Af + row * DD + k0;
        float4 x0 = *(const float4*)(p + kh);
        float4 x1 = *(const float4*)(p + kh + 4);
        float4 x2 = *(const float4*)(p + 16 + kh);
        float4 x3 = *(const float4*)(p + 16 + kh + 4);
        a[mt].w[0] = pkbf(x0.x, x0.y);  a[mt].w[1] = pkbf(x0.z, x0.w);
        a[mt].w[2] = pkbf(x1.x, x1.y);  a[mt].w[3] = pkbf(x1.z, x1.w);
        a[mt].w[4] = pkbf(x2.x, x2.y);  a[mt].w[5] = pkbf(x2.z, x2.w);
        a[mt].w[6] = pkbf(x3.x, x3.y);  a[mt].w[7] = pkbf(x3.z, x3.w);
      } else {
        const unsigned short* p = Ab + row * DD + k0;
        a[mt].q[0] = *(const uint4*)(p + kh);
        a[mt].q[1] = *(const uint4*)(p + 16 + kh);
      }
    }

    Frag bf[4];
#pragma unroll
    for (int nt = 0; nt < 4; ++nt) {
      const size_t col = (size_t)(n0 + nt * 16 + l16);
      const float* p   = W + col * DD + k0 + lhalf * 8;
      float4 x0 = *(const float4*)(p);
      float4 x1 = *(const float4*)(p + 4);
      float4 x2 = *(const float4*)(p + 16);
      float4 x3 = *(const float4*)(p + 20);
      bf[nt].w[0] = pkbf(x0.x, x0.y);  bf[nt].w[1] = pkbf(x0.z, x0.w);
      bf[nt].w[2] = pkbf(x1.x, x1.y);  bf[nt].w[3] = pkbf(x1.z, x1.w);
      bf[nt].w[4] = pkbf(x2.x, x2.y);  bf[nt].w[5] = pkbf(x2.z, x2.w);
      bf[nt].w[6] = pkbf(x3.x, x3.y);  bf[nt].w[7] = pkbf(x3.z, x3.w);
    }

#pragma unroll
    for (int mt = 0; mt < 2; ++mt)
#pragma unroll
      for (int nt = 0; nt < 4; ++nt)
        acc[mt][nt] = wmma_bf16(a[mt], bf[nt], acc[mt][nt]);
  }

  // C layout: VGPR r -> row M = r + 8*(lane>=16), col N = lane%16.
#pragma unroll
  for (int mt = 0; mt < 2; ++mt)
#pragma unroll
    for (int nt = 0; nt < 4; ++nt)
#pragma unroll
      for (int r = 0; r < 8; ++r) {
        const int   m = m0 + mt * 16 + r + lhalf * 8;
        const int   n = n0 + nt * 16 + l16;
        const float v = acc[mt][nt][r];
        if constexpr (QKVOUT) {
          const int bi = m >> 11, s = m & (SS - 1);
          const int h  = n >> 6,  hd = n & (HDD - 1);
          ((unsigned short*)Out)[(((size_t)(bi * HH + h)) * SS + s) * HDD + hd] =
              f2bf(v);
        } else {
          ((float*)Out)[(size_t)m * DD + n] = v;
        }
      }
}

// ---------------------------------------------------------------------------
// Flash attention. Grid (S/64, H, B), 128 threads = 4 waves; each wave owns a
// 16-row q tile; 32-key tiles per iteration; online softmax; 8 WMMA / iter.
// ---------------------------------------------------------------------------
__global__ __launch_bounds__(128) void flash_attn(
    const unsigned short* __restrict__ Qg, const unsigned short* __restrict__ Kg,
    const unsigned short* __restrict__ Vg, const int* __restrict__ mask,
    unsigned short* __restrict__ Og) {
  __shared__ __align__(16) unsigned short Vt[64][40];     // V^T tile [hd][key]
  __shared__ __align__(16) unsigned short Pl[4][16][40];  // per-wave P tile

  const int tid   = threadIdx.x;
  const int wave  = tid >> 5;
  const int lane  = tid & 31;
  const int lhalf = lane >> 4;
  const int l16   = lane & 15;
  const int b     = blockIdx.z, h = blockIdx.y;
  const int q0    = blockIdx.x * 64 + wave * 16;

  const size_t hoff = ((size_t)(b * HH + h)) * SS * HDD;
  const unsigned short* Q = Qg + hoff;
  const unsigned short* K = Kg + hoff;
  const unsigned short* V = Vg + hoff;
  const int* mrow = mask + (size_t)b * SS * SS;

  // Q fragments for k-steps 0 and 1 (HD=64 -> two K=32 slabs), kept resident.
  Frag qa[2];
  {
    const int row = q0 + l16;
    const int kh  = lhalf * 8;
#pragma unroll
    for (int ks = 0; ks < 2; ++ks) {
      const unsigned short* p = Q + (size_t)row * HDD + ks * 32;
      qa[ks].q[0] = *(const uint4*)(p + kh);
      qa[ks].q[1] = *(const uint4*)(p + 16 + kh);
    }
  }

  v8f o[4];
#pragma unroll
  for (int nt = 0; nt < 4; ++nt) o[nt] = zero8();
  float mi[8], li[8];
#pragma unroll
  for (int r = 0; r < 8; ++r) { mi[r] = -1e30f; li[r] = 0.0f; }

  for (int j0 = 0; j0 < SS; j0 += 32) {
    __syncthreads();  // previous iteration done reading LDS

    // Cooperative V stage: 32 keys x 64 hd -> bf16 transposed [hd][key].
    {
      const int kk = tid >> 2;
      const int hb = (tid & 3) * 16;
      Frag t;
      const unsigned short* p = V + (size_t)(j0 + kk) * HDD + hb;
      t.q[0] = *(const uint4*)(p);
      t.q[1] = *(const uint4*)(p + 8);
#pragma unroll
      for (int i = 0; i < 16; ++i) Vt[hb + i][kk] = t.u[i];
    }

    // Scores: two 16x16 C tiles over 32 keys, two K=32 steps each.
    v8f sc[2];
    sc[0] = zero8();
    sc[1] = zero8();
#pragma unroll
    for (int ct = 0; ct < 2; ++ct) {
#pragma unroll
      for (int ks = 0; ks < 2; ++ks) {
        Frag kb;
        const int key = j0 + ct * 16 + l16;
        const unsigned short* p = K + (size_t)key * HDD + ks * 32 + lhalf * 8;
        kb.q[0] = *(const uint4*)(p);
        kb.q[1] = *(const uint4*)(p + 16);
        sc[ct] = wmma_bf16(qa[ks], kb, sc[ct]);
      }
    }

    // Mask + online softmax. C row for slot r is q0 + r + 8*(lane>=16); the
    // 16 columns of a row live in lanes (l16), so shfl_xor 1..8 reduces rows.
#pragma unroll
    for (int r = 0; r < 8; ++r) {
      const int qrow = q0 + r + lhalf * 8;
#pragma unroll
      for (int ct = 0; ct < 2; ++ct) {
        const int key = j0 + ct * 16 + l16;
        float s = sc[ct][r] * kScale;
        if (mrow[(size_t)qrow * SS + key] == 0) s = kNeg;
        sc[ct][r] = s;
      }
      float mx = fmaxf(sc[0][r], sc[1][r]);
#pragma unroll
      for (int d = 1; d < 16; d <<= 1) mx = fmaxf(mx, __shfl_xor(mx, d, 32));
      const float nm   = fmaxf(mi[r], mx);
      const float corr = __expf(mi[r] - nm);
      mi[r] = nm;
      float rs = 0.0f;
#pragma unroll
      for (int ct = 0; ct < 2; ++ct) {
        const float pv = __expf(sc[ct][r] - nm);
        sc[ct][r] = pv;
        rs += pv;
      }
#pragma unroll
      for (int d = 1; d < 16; d <<= 1) rs += __shfl_xor(rs, d, 32);
      li[r] = li[r] * corr + rs;
#pragma unroll
      for (int nt = 0; nt < 4; ++nt) o[nt][r] *= corr;
    }

    // Stage P (16x32 bf16, A-fragment friendly row-major layout).
#pragma unroll
    for (int r = 0; r < 8; ++r)
#pragma unroll
      for (int ct = 0; ct < 2; ++ct)
        Pl[wave][r + lhalf * 8][ct * 16 + l16] = f2bf(sc[ct][r]);

    __syncthreads();  // Vt + Pl visible

    // O += P(16x32) @ V(32x64): one K=32 WMMA per 16-wide hd tile.
    Frag pa;
    {
      const int kh = lhalf * 8;
      pa.q[0] = *(const uint4*)&Pl[wave][l16][kh];
      pa.q[1] = *(const uint4*)&Pl[wave][l16][16 + kh];
    }
#pragma unroll
    for (int nt = 0; nt < 4; ++nt) {
      Frag vb;
      const int n   = nt * 16 + l16;
      const int kbs = lhalf * 8;
      vb.q[0] = *(const uint4*)&Vt[n][kbs];
      vb.q[1] = *(const uint4*)&Vt[n][kbs + 16];
      o[nt] = wmma_bf16(pa, vb, o[nt]);
    }
  }

  // Normalize and store to [B,S,D] (D index = h*64 + hd) as bf16.
#pragma unroll
  for (int r = 0; r < 8; ++r) {
    const float inv  = 1.0f / li[r];
    const int   qrow = q0 + r + lhalf * 8;
#pragma unroll
    for (int nt = 0; nt < 4; ++nt) {
      const int col = h * HDD + nt * 16 + l16;
      Og[((size_t)b * SS + qrow) * DD + col] = f2bf(o[nt][r] * inv);
    }
  }
}

// ---------------------------------------------------------------------------
// Host side. Inputs: query, key, value, mask, Wq, Wk, Wv, Wo.
// Scratch: Q/K bf16 live inside d_out (32 MB, dead before final GEMM writes);
// V bf16 + attention-out bf16 live in d_ws (32 MB needed).
// ---------------------------------------------------------------------------
extern "C" void kernel_launch(void* const* d_in, const int* in_sizes, int n_in,
                              void* d_out, int out_size, void* d_ws,
                              size_t ws_size, hipStream_t stream) {
  (void)in_sizes; (void)n_in; (void)out_size; (void)ws_size;
  const float* query = (const float*)d_in[0];
  const float* key   = (const float*)d_in[1];
  const float* value = (const float*)d_in[2];
  const int*   mask  = (const int*)d_in[3];
  const float* Wq    = (const float*)d_in[4];
  const float* Wk    = (const float*)d_in[5];
  const float* Wv    = (const float*)d_in[6];
  const float* Wo    = (const float*)d_in[7];

  const size_t NTOK = (size_t)BB * SS * DD;  // 8,388,608 elements
  unsigned short* qbf = (unsigned short*)d_out;          // 16 MB
  unsigned short* kbf = qbf + NTOK;                      // 16 MB
  unsigned short* vbf = (unsigned short*)d_ws;           // 16 MB
  unsigned short* abf = vbf + NTOK;                      // 16 MB

  dim3 ggrid(DD / 64, (BB * SS) / 32);  // (16, 256), 1 wave per block
  gemm_xwt<true, true><<<ggrid, 32, 0, stream>>>(query, Wq, qbf);
  gemm_xwt<true, true><<<ggrid, 32, 0, stream>>>(key,   Wk, kbf);
  gemm_xwt<true, true><<<ggrid, 32, 0, stream>>>(value, Wv, vbf);

  dim3 agrid(SS / 64, HH, BB);          // (32, 16, 4)
  flash_attn<<<agrid, 128, 0, stream>>>(qbf, kbf, vbf, mask, abf);

  gemm_xwt<false, false><<<ggrid, 32, 0, stream>>>(abf, Wo, (float*)d_out);
}
